// CLEAN_KAN_80384607912715
// MI455X (gfx1250) — compile-verified, matrix operand order checked by reading, not run on hardware
//
#include <hip/hip_runtime.h>
#include <hip/hip_bf16.h>

typedef __attribute__((ext_vector_type(16))) _Float16 v16h;
typedef __attribute__((ext_vector_type(8)))  _Float16 v8h;
typedef __attribute__((ext_vector_type(8)))  float    v8f;

#define B_SZ   16384
#define D_IN   1280
#define D_HID  512
#define D_OUT  229
#define NSLOT  7                   // 1 silu slot + 6 spline-basis slots per feature
#define K1     (D_IN * NSLOT)      // 8960
#define K2     (D_HID * NSLOT)     // 3584
#define N2PAD  256                 // layer-2 N padded to 2 x 128 tiles

#define MT  128
#define NT  128
#define KC  64
#define LDT 72                     // LDS row stride (elements), 144B: conflict-free b128

// LDS layout inside dynamic smem (element offsets): [A buf0][A buf1][W buf0][W buf1]
#define A_OFF(buf) ((buf) * (MT * LDT))
#define W_OFF(buf) (2 * MT * LDT + (buf) * (NT * LDT))

// ---------------------------------------------------------------------------
// CDNA5 async global->LDS copy (16 bytes per lane), tracked by ASYNCcnt.
// LDS destination operand = workgroup-relative LDS byte address = low 32 bits
// of the flat shared-aperture pointer.
// ---------------------------------------------------------------------------
__device__ __forceinline__ void async_copy_b128(const _Float16* g, _Float16* l) {
    unsigned           lds = (unsigned)(unsigned long long)(uintptr_t)l;
    unsigned long long ga  = (unsigned long long)(uintptr_t)g;
    asm volatile("global_load_async_to_lds_b128 %0, %1, off"
                 :: "v"(lds), "v"(ga)
                 : "memory");
}

__device__ __forceinline__ void wait_async_le8() {
    asm volatile("s_wait_asynccnt 0x8" ::: "memory");
}
__device__ __forceinline__ void wait_async_0() {
    asm volatile("s_wait_asynccnt 0x0" ::: "memory");
}

// ---------------------------------------------------------------------------
// KAN activation: silu + cubic B-spline bases on uniform knots t_j = -3 + j*2/3
// ---------------------------------------------------------------------------
__device__ __forceinline__ float kan_knot(int j) {
    return -3.0f + (float)j * (2.0f / 3.0f);
}

__device__ __forceinline__ void kan_basis(float x, float* out /*[7]*/) {
    out[0] = x / (1.0f + __expf(-x));          // silu
    float b[9];
#pragma unroll
    for (int j = 0; j < 9; ++j)
        b[j] = (x >= kan_knot(j) && x < kan_knot(j + 1)) ? 1.0f : 0.0f;
#pragma unroll
    for (int k = 1; k <= 3; ++k) {
        float inv = 1.0f / ((float)k * (2.0f / 3.0f));
#pragma unroll
        for (int j = 0; j + k < 9; ++j) {
            float left  = (x - kan_knot(j)) * inv;
            float right = (kan_knot(j + k + 1) - x) * inv;
            b[j] = left * b[j] + right * b[j + 1];
        }
    }
#pragma unroll
    for (int j = 0; j < 6; ++j) out[1 + j] = b[j];
}

// ---------------------------------------------------------------------------
// Weight prep: interleave [base | spline*scaler] into f16, K-major slots of 7
// ---------------------------------------------------------------------------
__global__ void kan_prep_w1(const float* __restrict__ base_w,
                            const float* __restrict__ spline_w,
                            const float* __restrict__ scaler,
                            _Float16* __restrict__ W) {
    int idx = blockIdx.x * blockDim.x + threadIdx.x;
    if (idx >= D_HID * D_IN) return;
    float sc = scaler[idx];
    _Float16* w = W + (size_t)idx * NSLOT;
    w[0] = (_Float16)base_w[idx];
#pragma unroll
    for (int j = 0; j < 6; ++j)
        w[1 + j] = (_Float16)(spline_w[(size_t)idx * 6 + j] * sc);
}

__global__ void kan_prep_w2(const float* __restrict__ base_w,
                            const float* __restrict__ spline_w,
                            const float* __restrict__ scaler,
                            _Float16* __restrict__ W) {
    int idx = blockIdx.x * blockDim.x + threadIdx.x;   // over N2PAD * D_HID
    if (idx >= N2PAD * D_HID) return;
    int o = idx / D_HID;
    int i = idx - o * D_HID;
    _Float16* w = W + (size_t)idx * NSLOT;
    if (o < D_OUT) {
        int src = o * D_HID + i;
        float sc = scaler[src];
        w[0] = (_Float16)base_w[src];
#pragma unroll
        for (int j = 0; j < 6; ++j)
            w[1 + j] = (_Float16)(spline_w[(size_t)src * 6 + j] * sc);
    } else {
#pragma unroll
        for (int j = 0; j < NSLOT; ++j) w[j] = (_Float16)0.0f;
    }
}

// ---------------------------------------------------------------------------
// Layer-1 augmented activations: x (B x D_IN) -> A1 (B x K1) f16
// ---------------------------------------------------------------------------
__global__ void kan_act1(const float* __restrict__ x, _Float16* __restrict__ A) {
    int idx = blockIdx.x * blockDim.x + threadIdx.x;
    if (idx >= B_SZ * D_IN) return;
    float v[NSLOT];
    kan_basis(x[idx], v);
    _Float16* a = A + (size_t)idx * NSLOT;
#pragma unroll
    for (int j = 0; j < NSLOT; ++j) a[j] = (_Float16)v[j];
}

// ---------------------------------------------------------------------------
// LayerNorm(512) fused with layer-2 augmented activation build
// one block (256 thr) per batch row; each thread owns 2 hidden columns
// ---------------------------------------------------------------------------
__global__ void kan_ln_act2(const float* __restrict__ H,
                            const float* __restrict__ gamma,
                            const float* __restrict__ beta,
                            _Float16* __restrict__ A) {
    __shared__ float red[256];
    const int row = blockIdx.x;
    const int tid = threadIdx.x;
    const float* h = H + (size_t)row * D_HID;
    float v0 = h[tid];
    float v1 = h[tid + 256];

    red[tid] = v0 + v1;
    __syncthreads();
    for (int s = 128; s > 0; s >>= 1) {
        if (tid < s) red[tid] += red[tid + s];
        __syncthreads();
    }
    float mu = red[0] * (1.0f / (float)D_HID);
    __syncthreads();

    float d0 = v0 - mu, d1 = v1 - mu;
    red[tid] = d0 * d0 + d1 * d1;
    __syncthreads();
    for (int s = 128; s > 0; s >>= 1) {
        if (tid < s) red[tid] += red[tid + s];
        __syncthreads();
    }
    float rstd = rsqrtf(red[0] * (1.0f / (float)D_HID) + 1e-5f);

    float g0 = d0 * rstd * gamma[tid]       + beta[tid];
    float g1 = d1 * rstd * gamma[tid + 256] + beta[tid + 256];

    float v[NSLOT];
    kan_basis(g0, v);
    _Float16* a = A + (size_t)row * K2 + (size_t)tid * NSLOT;
#pragma unroll
    for (int j = 0; j < NSLOT; ++j) a[j] = (_Float16)v[j];
    kan_basis(g1, v);
    a = A + (size_t)row * K2 + (size_t)(tid + 256) * NSLOT;
#pragma unroll
    for (int j = 0; j < NSLOT; ++j) a[j] = (_Float16)v[j];
}

// ---------------------------------------------------------------------------
// f16 -> f32 WMMA GEMM:  C(MxN) = A(MxK, row-major) * W(NxK, row-major)^T
// block = 256 thr = 8 waves; tile 128x128; KC=64 staging, LDS double-buffered,
// fed by global_load_async_to_lds_b128 pipelined two stages deep (ASYNCcnt).
// wave grid 2(M) x 4(N); wave tile 64x32 = 4x2 wmma tiles of 16x16
// ---------------------------------------------------------------------------
__global__ __launch_bounds__(256)
void kan_gemm(const _Float16* __restrict__ A, const _Float16* __restrict__ W,
              float* __restrict__ C, int K, int ldc, int nLimit) {
    extern __shared__ __align__(16) _Float16 smem[];

    const int tid  = threadIdx.x;
    const int lane = tid & 31;
    const int wid  = tid >> 5;          // 0..7
    const int wm   = (wid & 1) * 64;    // wave M offset in tile
    const int wn   = (wid >> 1) * 32;   // wave N offset in tile
    const int hf   = lane >> 4;         // half-wave select (K-interleave)
    const int r    = lane & 15;         // row within 16 (M for A, N for B)

    const int bm = blockIdx.x * MT;
    const int bn = blockIdx.y * NT;

    // staging coordinates: 1024 slots of 8 f16, 4 per thread
    int srow[4], sseg[4];
#pragma unroll
    for (int t = 0; t < 4; ++t) {
        int slot = tid + t * 256;
        srow[t]  = slot >> 3;
        sseg[t]  = (slot & 7) * 8;
    }

    const _Float16* Abase = A + (size_t)bm * K;
    const _Float16* Wbase = W + (size_t)bn * K;

    auto issue_stage = [&](int buf, int kc) {
#pragma unroll
        for (int t = 0; t < 4; ++t) {
            async_copy_b128(Abase + (size_t)srow[t] * K + kc + sseg[t],
                            &smem[A_OFF(buf) + srow[t] * LDT + sseg[t]]);
            async_copy_b128(Wbase + (size_t)srow[t] * K + kc + sseg[t],
                            &smem[W_OFF(buf) + srow[t] * LDT + sseg[t]]);
        }
    };

    v8f acc[4][2];
#pragma unroll
    for (int i = 0; i < 4; ++i)
#pragma unroll
        for (int j = 0; j < 2; ++j)
#pragma unroll
            for (int e = 0; e < 8; ++e) acc[i][j][e] = 0.0f;

    const int nstage = K / KC;          // 140 or 56 -> always >= 2
    issue_stage(0, 0);
    issue_stage(1, KC);

    for (int s = 0; s < nstage; ++s) {
        // 8 async issues per thread per stage; in-order completion =>
        // waiting <=8 retires stage s while stage s+1 stays in flight.
        if (s + 1 < nstage) wait_async_le8();
        else                wait_async_0();
        __syncthreads();

        const int buf = s & 1;
        const _Float16* Asb = &smem[A_OFF(buf)];
        const _Float16* Wsb = &smem[W_OFF(buf)];

#pragma unroll
        for (int kk = 0; kk < KC; kk += 32) {
            v16h af[4], bfr[2];
#pragma unroll
            for (int mt = 0; mt < 4; ++mt) {
                const _Float16* p = &Asb[(wm + mt * 16 + r) * LDT + kk + hf * 8];
                v8h lo = *(const v8h*)p;
                v8h hi = *(const v8h*)(p + 16);
#pragma unroll
                for (int e = 0; e < 8; ++e) { af[mt][e] = lo[e]; af[mt][8 + e] = hi[e]; }
            }
#pragma unroll
            for (int nt = 0; nt < 2; ++nt) {
                const _Float16* p = &Wsb[(wn + nt * 16 + r) * LDT + kk + hf * 8];
                v8h lo = *(const v8h*)p;
                v8h hi = *(const v8h*)(p + 16);
#pragma unroll
                for (int e = 0; e < 8; ++e) { bfr[nt][e] = lo[e]; bfr[nt][8 + e] = hi[e]; }
            }
#pragma unroll
            for (int mt = 0; mt < 4; ++mt)
#pragma unroll
                for (int nt = 0; nt < 2; ++nt)
                    acc[mt][nt] = __builtin_amdgcn_wmma_f32_16x16x32_f16(
                        false, af[mt], false, bfr[nt], (short)0, acc[mt][nt],
                        false, false);
        }
        __syncthreads();                 // everyone done reading buf before refill
        if (s + 2 < nstage) issue_stage(buf, (s + 2) * KC);
    }

    // epilogue: D layout -> lane r = N column, vgpr e = M row (+8 for hi half)
#pragma unroll
    for (int mt = 0; mt < 4; ++mt) {
#pragma unroll
        for (int nt = 0; nt < 2; ++nt) {
            int n = bn + wn + nt * 16 + r;
            if (n < nLimit) {
                int mbase = bm + wm + mt * 16 + hf * 8;
#pragma unroll
                for (int e = 0; e < 8; ++e)
                    C[(size_t)(mbase + e) * ldc + n] = acc[mt][nt][e];
            }
        }
    }
}

// ---------------------------------------------------------------------------
extern "C" void kernel_launch(void* const* d_in, const int* in_sizes, int n_in,
                              void* d_out, int out_size, void* d_ws, size_t ws_size,
                              hipStream_t stream) {
    const float* x         = (const float*)d_in[0];
    const float* base_w1   = (const float*)d_in[1];
    const float* spline_w1 = (const float*)d_in[2];
    const float* scaler1   = (const float*)d_in[3];
    const float* ln_gamma  = (const float*)d_in[4];
    const float* ln_beta   = (const float*)d_in[5];
    const float* base_w2   = (const float*)d_in[6];
    const float* spline_w2 = (const float*)d_in[7];
    const float* scaler2   = (const float*)d_in[8];
    float* out = (float*)d_out;

    char*  ws  = (char*)d_ws;
    size_t off = 0;
    auto alloc = [&](size_t bytes) -> void* {
        void* p = ws + off;
        off += (bytes + 255) & ~(size_t)255;
        return p;
    };
    _Float16* W1 = (_Float16*)alloc((size_t)D_HID  * K1 * sizeof(_Float16));
    _Float16* W2 = (_Float16*)alloc((size_t)N2PAD  * K2 * sizeof(_Float16));
    _Float16* A1 = (_Float16*)alloc((size_t)B_SZ   * K1 * sizeof(_Float16));
    float*    H  = (float*)   alloc((size_t)B_SZ   * D_HID * sizeof(float));
    _Float16* A2 = (_Float16*)alloc((size_t)B_SZ   * K2 * sizeof(_Float16));

    const size_t gemm_lds = (size_t)2 * (MT + NT) * LDT * sizeof(_Float16); // 73728 B

    kan_prep_w1<<<(D_HID * D_IN + 255) / 256, 256, 0, stream>>>(base_w1, spline_w1, scaler1, W1);
    kan_prep_w2<<<(N2PAD * D_HID + 255) / 256, 256, 0, stream>>>(base_w2, spline_w2, scaler2, W2);
    kan_act1<<<(B_SZ * D_IN + 255) / 256, 256, 0, stream>>>(x, A1);

    kan_gemm<<<dim3(B_SZ / MT, D_HID / NT), 256, gemm_lds, stream>>>(A1, W1, H, K1, D_HID, D_HID);
    kan_ln_act2<<<B_SZ, 256, 0, stream>>>(H, ln_gamma, ln_beta, A2);
    kan_gemm<<<dim3(B_SZ / MT, N2PAD / NT), 256, gemm_lds, stream>>>(A2, W2, out, K2, D_OUT, D_OUT);
}
